// JAttention_59579786330807
// MI455X (gfx1250) — matrix operand order repaired and checked by reading, not compiled, not run on hardware
//
#include <hip/hip_runtime.h>

typedef _Float16 half8  __attribute__((ext_vector_type(8)));
typedef _Float16 half16 __attribute__((ext_vector_type(16)));
typedef float    float8 __attribute__((ext_vector_type(8)));

#define WAVES 16
#define TPB   512
#define RT    5          // 80 rows = 5 row-tiles of 16 (68 real rows + pad)
#define PSA   272        // LDS stride (halves) for 256-wide buffers
#define PSQ   784        // LDS stride for qkv buffer (768 cols + pad)
#define PSM   528        // LDS stride for fc1 output (512 cols + pad)

// LDS layout (bytes)
#define OFF_A   0                      // 80 x 272 f16 = 43520
#define OFF_B   43520                  // 80 x 272 f16 = 43520
#define OFF_Q   87040                  // 96 x 784 f16 = 150528
#define OFF_S   237568                 // 16 waves x 32x20 f32 = 40960
#define OFF_P   278528                 // 16 waves x 32x32 f16 = 32768
#define LDS_TOTAL 311296

// workspace offsets (halves) for f16-transposed weights  Wt[n][k]
#define WT_L    0          // 256x256
#define WT_QKV  65536      // 768x256
#define WT_PROJ 262144     // 256x256
#define WT_FC1  327680     // 512x256
#define WT_FC2  458752     // 256x512
#define WT_TOTAL 589824

// ---------------- JAX threefry2x32 reproduction (key = [0, 42]) -------------
__device__ __forceinline__ unsigned rotl32(unsigned x, int r) {
  return (x << r) | (x >> (32 - r));
}
__device__ __forceinline__ void tf_round(unsigned& x0, unsigned& x1, int r) {
  x0 += x1; x1 = rotl32(x1, r); x1 ^= x0;
}
__device__ __forceinline__ void threefry2x32(unsigned c0, unsigned c1,
                                             unsigned& o0, unsigned& o1) {
  const unsigned k0 = 0u, k1 = 42u;
  const unsigned k2 = k0 ^ k1 ^ 0x1BD11BDAu;
  unsigned x0 = c0 + k0, x1 = c1 + k1;
  tf_round(x0,x1,13); tf_round(x0,x1,15); tf_round(x0,x1,26); tf_round(x0,x1,6);
  x0 += k1; x1 += k2 + 1u;
  tf_round(x0,x1,17); tf_round(x0,x1,29); tf_round(x0,x1,16); tf_round(x0,x1,24);
  x0 += k2; x1 += k0 + 2u;
  tf_round(x0,x1,13); tf_round(x0,x1,15); tf_round(x0,x1,26); tf_round(x0,x1,6);
  x0 += k0; x1 += k1 + 3u;
  tf_round(x0,x1,17); tf_round(x0,x1,29); tf_round(x0,x1,16); tf_round(x0,x1,24);
  x0 += k1; x1 += k2 + 4u;
  tf_round(x0,x1,13); tf_round(x0,x1,15); tf_round(x0,x1,26); tf_round(x0,x1,6);
  x0 += k2; x1 += k0 + 5u;
  o0 = x0; o1 = x1;
}
// bernoulli(key(42), 0.15) over flat attn index (B*8*17*17 elements, even)
__device__ __forceinline__ bool dropmask(unsigned idx) {
  const unsigned HALF = 18939904u;   // (16384*8*17*17)/2
  bool second = idx >= HALF;
  unsigned c0 = second ? (idx - HALF) : idx;
  unsigned c1 = c0 + HALF;
  unsigned o0, o1;
  threefry2x32(c0, c1, o0, o1);
  unsigned bits = second ? o1 : o0;
  float u = __uint_as_float((bits >> 9) | 0x3F800000u) - 1.0f;
  return u < 0.15f;
}

// ---------------- generic LDS-A x global-f16-Wt GEMM via WMMA ---------------
template<int K, int NOUT, int LDA, int LDC, bool GELU>
__device__ __forceinline__ void gemm(const _Float16* __restrict__ A,
                                     const _Float16* __restrict__ Wt,
                                     const float* __restrict__ bias,
                                     _Float16* __restrict__ C,
                                     int wave, int lane) {
  const int lh = lane >> 4;      // K-half selector
  const int nl = lane & 15;      // A-row m / B-col n within tile
  const float8 fzero = {0.f,0.f,0.f,0.f,0.f,0.f,0.f,0.f};
  for (int nt = wave; nt < (NOUT >> 4); nt += WAVES) {
    float8 acc[RT];
#pragma unroll
    for (int r = 0; r < RT; ++r) acc[r] = fzero;
    float bv = bias ? bias[nt * 16 + nl] : 0.0f;
    for (int kb = 0; kb < K; kb += 32) {
      // B fragment: 16 consecutive K from transposed weight row (32B load)
      half16 bf = *(const half16*)(Wt + (size_t)(nt * 16 + nl) * K + kb + lh * 16);
#pragma unroll
      for (int r = 0; r < RT; ++r) {
        const _Float16* ap = A + (r * 16 + nl) * LDA + kb + lh * 16;
        half8 alo = *(const half8*)ap;
        half8 ahi = *(const half8*)(ap + 8);
        half16 af = __builtin_shufflevector(alo, ahi,
            0,1,2,3,4,5,6,7,8,9,10,11,12,13,14,15);
        acc[r] = __builtin_amdgcn_wmma_f32_16x16x32_f16(
            false, af, false, bf, (short)0, acc[r], false, false);
      }
    }
#pragma unroll
    for (int r = 0; r < RT; ++r) {
#pragma unroll
      for (int j = 0; j < 8; ++j) {
        float v = acc[r][j] + bv;
        if (GELU) v = 0.5f * v * (1.0f + erff(v * 0.70710678118654752f));
        C[(r * 16 + lh * 8 + j) * LDC + nt * 16 + nl] = (_Float16)v;
      }
    }
  }
}

// ---------------- per-(sample,head) attention -------------------------------
__device__ __forceinline__ void attention(const _Float16* __restrict__ bufQ,
                                          _Float16* __restrict__ bufO,
                                          float* __restrict__ sS,
                                          _Float16* __restrict__ sP,
                                          int wave, int lane, int blockBase) {
  const int lh = lane >> 4, nl = lane & 15;
  const float8 fzero = {0.f,0.f,0.f,0.f,0.f,0.f,0.f,0.f};
  float* Sw = sS + wave * (32 * 20);
  _Float16* Pw = sP + wave * (32 * 32);
  for (int inst = wave; inst < 32; inst += WAVES) {
    int s = inst >> 3, h = inst & 7;
    int rbase = s * 17;
    const _Float16* q = bufQ + (size_t)rbase * PSQ + h * 32;
    const _Float16* k = bufQ + (size_t)rbase * PSQ + 256 + h * 32;
    const _Float16* v = bufQ + (size_t)rbase * PSQ + 512 + h * 32;

    // S = q @ k^T   (K = head_dim = 32 exactly)
#pragma unroll
    for (int mt = 0; mt < 2; ++mt) {
      const _Float16* ap = q + (mt * 16 + nl) * PSQ + lh * 16;
      half8 alo = *(const half8*)ap;
      half8 ahi = *(const half8*)(ap + 8);
      half16 af = __builtin_shufflevector(alo, ahi,
          0,1,2,3,4,5,6,7,8,9,10,11,12,13,14,15);
#pragma unroll
      for (int nt = 0; nt < 2; ++nt) {
        half16 bf = *(const half16*)(k + (nt * 16 + nl) * PSQ + lh * 16);
        float8 acc = fzero;
        acc = __builtin_amdgcn_wmma_f32_16x16x32_f16(
            false, af, false, bf, (short)0, acc, false, false);
        int col = nt * 16 + nl;
        if (col < 17) {
#pragma unroll
          for (int j = 0; j < 8; ++j)
            Sw[(mt * 16 + lh * 8 + j) * 20 + col] = acc[j];
        }
      }
    }

    // softmax + DropKey mask (one row per lane)
    int i = lane;
    if (i < 17) {
      unsigned base = ((unsigned)(((blockBase + s) * 8 + h) * 17 + i)) * 17u;
      float vals[17];
      float mx = -3.0e38f;
#pragma unroll
      for (int j = 0; j < 17; ++j) {
        float sv = Sw[i * 20 + j] * 0.17677669529663687f;  // 32^-0.5
        if (dropmask(base + (unsigned)j)) sv -= 1.0e12f;
        vals[j] = sv;
        mx = fmaxf(mx, sv);
      }
      float sum = 0.0f;
#pragma unroll
      for (int j = 0; j < 17; ++j) { vals[j] = __expf(vals[j] - mx); sum += vals[j]; }
      float inv = 1.0f / sum;
#pragma unroll
      for (int j = 0; j < 17; ++j) Pw[i * 32 + j] = (_Float16)(vals[j] * inv);
#pragma unroll
      for (int j = 17; j < 32; ++j) Pw[i * 32 + j] = (_Float16)0.f;
    } else {
#pragma unroll
      for (int j = 0; j < 32; ++j) Pw[i * 32 + j] = (_Float16)0.f;
    }

    // O = P @ v  (K = 32 tokens, P cols >=17 zeroed, v pad rows zeroed)
#pragma unroll
    for (int mt = 0; mt < 2; ++mt) {
      const _Float16* ap = Pw + (mt * 16 + nl) * 32 + lh * 16;
      half8 alo = *(const half8*)ap;
      half8 ahi = *(const half8*)(ap + 8);
      half16 af = __builtin_shufflevector(alo, ahi,
          0,1,2,3,4,5,6,7,8,9,10,11,12,13,14,15);
#pragma unroll
      for (int nt = 0; nt < 2; ++nt) {
        half16 bf;
#pragma unroll
        for (int e = 0; e < 16; ++e)
          bf[e] = v[(lh * 16 + e) * PSQ + nt * 16 + nl];   // column of v
        float8 acc = fzero;
        acc = __builtin_amdgcn_wmma_f32_16x16x32_f16(
            false, af, false, bf, (short)0, acc, false, false);
#pragma unroll
        for (int j = 0; j < 8; ++j) {
          int tok = mt * 16 + lh * 8 + j;
          if (tok < 17)
            bufO[(rbase + tok) * PSA + h * 32 + nt * 16 + nl] = (_Float16)acc[j];
        }
      }
    }
  }
}

// ---------------- weight prep: fp32 -> f16, transposed to Wt[n][k] ----------
__global__ void prep_weights(const float* __restrict__ Lw,
                             const float* __restrict__ Qw,
                             const float* __restrict__ Pjw,
                             const float* __restrict__ F1,
                             const float* __restrict__ F2,
                             _Float16* __restrict__ wt) {
  unsigned i = blockIdx.x * blockDim.x + threadIdx.x;
  if (i < 65536u)  { unsigned n = i >> 8, kk = i & 255u; wt[WT_L    + i] = (_Float16)Lw [kk * 256u + n]; return; }
  i -= 65536u;
  if (i < 196608u) { unsigned n = i >> 8, kk = i & 255u; wt[WT_QKV  + i] = (_Float16)Qw [kk * 768u + n]; return; }
  i -= 196608u;
  if (i < 65536u)  { unsigned n = i >> 8, kk = i & 255u; wt[WT_PROJ + i] = (_Float16)Pjw[kk * 256u + n]; return; }
  i -= 65536u;
  if (i < 131072u) { unsigned n = i >> 8, kk = i & 255u; wt[WT_FC1  + i] = (_Float16)F1 [kk * 512u + n]; return; }
  i -= 131072u;
  if (i < 131072u) { unsigned n = i >> 9, kk = i & 511u; wt[WT_FC2  + i] = (_Float16)F2 [kk * 256u + n]; return; }
}

// ---------------- fully fused transformer block (4 samples / workgroup) -----
__global__ __launch_bounds__(TPB, 1)
void fused_block(const float* __restrict__ x,
                 const float* __restrict__ emb_w, const float* __restrict__ emb_b,
                 const float* __restrict__ pa,    const float* __restrict__ L_b,
                 const float* __restrict__ proj_b,
                 const float* __restrict__ fc1_b, const float* __restrict__ fc2_b,
                 const float* __restrict__ fc_w,  const float* __restrict__ fc_b,
                 const _Float16* __restrict__ wt, float* __restrict__ out) {
  extern __shared__ char smem[];
  _Float16* bufA = (_Float16*)(smem + OFF_A);
  _Float16* bufB = (_Float16*)(smem + OFF_B);
  _Float16* bufQ = (_Float16*)(smem + OFF_Q);
  float*    sS   = (float*)   (smem + OFF_S);
  _Float16* sP   = (_Float16*)(smem + OFF_P);
  _Float16* bufM = bufQ;   // fc1 output aliases dead qkv buffer

  const int tid = threadIdx.x;
  const int wave = tid >> 5, lane = tid & 31;
  const int wg = blockIdx.x;

  // stage0: embed + pos  -> bufA ; zero qkv pad rows (80..95)
  for (int idx = tid; idx < 80 * 256; idx += TPB) {
    int row = idx >> 8, c = idx & 255;
    float hv = 0.0f;
    if (row < 68) {
      int grow = wg * 68 + row;
      int t = row % 17;
      float x0 = x[grow * 2 + 0], x1 = x[grow * 2 + 1];
      hv = x0 * emb_w[c] + x1 * emb_w[256 + c] + emb_b[c] + pa[t * 256 + c];
    }
    bufA[row * PSA + c] = (_Float16)hv;
  }
  for (int idx = tid; idx < 16 * PSQ; idx += TPB)
    bufQ[80 * PSQ + idx] = (_Float16)0.f;
  __syncthreads();

  // stage1: h1 = h0 @ L_w + L_b
  gemm<256, 256, PSA, PSA, false>(bufA, wt + WT_L, L_b, bufB, wave, lane);
  __syncthreads();
  // stage2: qkv = h1 @ qkv_w   (no bias)
  gemm<256, 768, PSA, PSQ, false>(bufB, wt + WT_QKV, nullptr, bufQ, wave, lane);
  __syncthreads();
  // stage3: attention -> bufA
  attention(bufQ, bufA, sS, sP, wave, lane, wg * 4);
  __syncthreads();
  // stage4: p = o @ proj_w + proj_b
  gemm<256, 256, PSA, PSA, false>(bufA, wt + WT_PROJ, proj_b, bufB, wave, lane);
  __syncthreads();
  // stage5: m = gelu(p @ fc1_w + fc1_b)
  gemm<256, 512, PSA, PSM, true>(bufB, wt + WT_FC1, fc1_b, bufM, wave, lane);
  __syncthreads();
  // stage6: m2 = m @ fc2_w + fc2_b
  gemm<512, 256, PSM, PSA, false>(bufM, wt + WT_FC2, fc2_b, bufA, wave, lane);
  __syncthreads();
  // stage7: out = m2 @ fc_w + fc_b   (Nout=2, scalar dot products)
  for (int idx = tid; idx < 68 * 2; idx += TPB) {
    int row = idx >> 1, c = idx & 1;
    float s = fc_b[c];
    for (int k2 = 0; k2 < 256; ++k2)
      s += (float)bufA[row * PSA + k2] * fc_w[k2 * 2 + c];
    out[(wg * 68 + row) * 2 + c] = s;
  }
}

extern "C" void kernel_launch(void* const* d_in, const int* in_sizes, int n_in,
                              void* d_out, int out_size, void* d_ws, size_t ws_size,
                              hipStream_t stream) {
  const float* x      = (const float*)d_in[0];
  const float* emb_w  = (const float*)d_in[1];
  const float* emb_b  = (const float*)d_in[2];
  const float* pa     = (const float*)d_in[3];
  const float* L_w    = (const float*)d_in[4];
  const float* L_b    = (const float*)d_in[5];
  const float* qkv_w  = (const float*)d_in[6];
  const float* proj_w = (const float*)d_in[7];
  const float* proj_b = (const float*)d_in[8];
  const float* fc1_w  = (const float*)d_in[9];
  const float* fc1_b  = (const float*)d_in[10];
  const float* fc2_w  = (const float*)d_in[11];
  const float* fc2_b  = (const float*)d_in[12];
  const float* fc_w   = (const float*)d_in[13];
  const float* fc_b   = (const float*)d_in[14];
  _Float16* wt = (_Float16*)d_ws;

  prep_weights<<<WT_TOTAL / 256, 256, 0, stream>>>(L_w, qkv_w, proj_w, fc1_w, fc2_w, wt);
  fused_block<<<16384 / 4, TPB, LDS_TOTAL, stream>>>(
      x, emb_w, emb_b, pa, L_b, proj_b, fc1_b, fc2_b, fc_w, fc_b, wt,
      (float*)d_out);
}
